// MixtralSparseMoeBlock_34067680592359
// MI455X (gfx1250) — matrix operand order repaired and checked by reading, not compiled, not run on hardware
//
#include <hip/hip_runtime.h>
#include <hip/hip_bf16.h>
#include <math.h>

// Problem dims (fixed by reference)
#define BB 2
#define SS 2048
#define TT (BB * SS)   // 4096 tokens
#define HH 1024
#define II 2048
#define EE 8
#define TOPK 2

typedef __bf16 bf16_t;
typedef __attribute__((ext_vector_type(16))) __bf16 v16bf;
typedef __attribute__((ext_vector_type(8)))  __bf16 v8bf;
typedef __attribute__((ext_vector_type(8)))  float  v8f;

#define TILE_M 32
#define TILES_PER_E (TT / TILE_M)   // 128
#define XS_STRIDE 1032              // bf16 elems; 2064B row stride (16B aligned, +4 dword skew)
#define HD_STRIDE 2056              // bf16 elems; 4112B row stride (16B aligned, +4 dword skew)

static __device__ __forceinline__ v8f wmma_bf16(v16bf a, v16bf b, v8f c) {
    return __builtin_amdgcn_wmma_f32_16x16x32_bf16(false, a, false, b, (short)0, c,
                                                   false, false);
}

// Fast reciprocal (v_rcp_f32, ~1 ULP) — avoids the IEEE div expansion.
static __device__ __forceinline__ float fast_rcp(float x) {
    return __builtin_amdgcn_rcpf(x);
}

// silu(g) = g * sigmoid(g), with v_exp + v_rcp (no div_scale chains).
static __device__ __forceinline__ float silu(float g) {
    return g * fast_rcp(1.f + __expf(-g));
}

// A fragment (16x32 bf16) from an LDS tile, rows of `stride` bf16 elements.
// lane<16:  row=lane,    K elems {kb+0..7, kb+16..23}
// lane>=16: row=lane-16, K elems {kb+8..15, kb+24..31}
static __device__ __forceinline__ v16bf load_a_frag(const bf16_t* base, int stride,
                                                    int m, int kb, int half) {
    const bf16_t* p = base + m * stride + kb + half;
    v8bf c0 = *(const v8bf*)(p);
    v8bf c1 = *(const v8bf*)(p + 16);
    return __builtin_shufflevector(c0, c1, 0, 1, 2, 3, 4, 5, 6, 7,
                                   8, 9, 10, 11, 12, 13, 14, 15);
}

// B fragment (32x16): lane <-> N column (lane&15); K rows {8h+0..7, 16+8h+0..7} strided fp32 -> bf16.
static __device__ __forceinline__ v16bf load_b_frag(const float* p, int stride) {
    v16bf b;
#pragma unroll
    for (int s = 0; s < 16; ++s)
        b[s] = (bf16_t)p[(size_t)((s < 8) ? s : s + 8) * stride];
    return b;
}

__global__ void zero_kernel(float* __restrict__ out, int* __restrict__ counts) {
    size_t i = (size_t)blockIdx.x * blockDim.x + threadIdx.x;
    ((float4*)out)[i] = make_float4(0.f, 0.f, 0.f, 0.f);
    if (blockIdx.x == 0 && threadIdx.x < EE) counts[threadIdx.x] = 0;
}

// One wave32 per token: logits, softmax, top-2, renorm, append to expert lists.
__global__ __launch_bounds__(256) void router_kernel(
    const float* __restrict__ x, const float* __restrict__ rw,
    float* __restrict__ logits_out, int* __restrict__ counts,
    int* __restrict__ tok_list, float* __restrict__ w_list) {
    int lane = threadIdx.x & 31;
    int wave = threadIdx.x >> 5;
    int t = blockIdx.x * 8 + wave;

    float acc[EE];
#pragma unroll
    for (int e = 0; e < EE; ++e) acc[e] = 0.f;
    const float* xr = x + (size_t)t * HH;
    for (int h = lane; h < HH; h += 32) {
        float xv = xr[h];
        const float* r = rw + (size_t)h * EE;
#pragma unroll
        for (int e = 0; e < EE; ++e) acc[e] = fmaf(xv, r[e], acc[e]);
    }
#pragma unroll
    for (int off = 16; off > 0; off >>= 1) {
#pragma unroll
        for (int e = 0; e < EE; ++e) acc[e] += __shfl_xor(acc[e], off, 32);
    }
    if (lane == 0) {
        float m = acc[0];
#pragma unroll
        for (int e = 1; e < EE; ++e) m = fmaxf(m, acc[e]);
        float p[EE], s = 0.f;
#pragma unroll
        for (int e = 0; e < EE; ++e) { p[e] = __expf(acc[e] - m); s += p[e]; }
        float inv = fast_rcp(s);
#pragma unroll
        for (int e = 0; e < EE; ++e) {
            p[e] *= inv;
            logits_out[(size_t)t * EE + e] = acc[e];
        }
        int i1 = 0;
#pragma unroll
        for (int e = 1; e < EE; ++e) i1 = (p[e] > p[i1]) ? e : i1;
        int i2 = (i1 == 0) ? 1 : 0;
#pragma unroll
        for (int e = 0; e < EE; ++e)
            if (e != i1 && p[e] > p[i2]) i2 = e;
        float rinv = fast_rcp(p[i1] + p[i2]);
        int s1 = atomicAdd(&counts[i1], 1);
        tok_list[i1 * TT + s1] = t;
        w_list[i1 * TT + s1] = p[i1] * rinv;
        int s2 = atomicAdd(&counts[i2], 1);
        tok_list[i2 * TT + s2] = t;
        w_list[i2 * TT + s2] = p[i2] * rinv;
    }
}

// Fused per-expert SwiGLU FFN for a tile of 32 routed tokens.
__global__ __launch_bounds__(256) void moe_expert_kernel(
    const float* __restrict__ x, const float* __restrict__ gate_w,
    const float* __restrict__ up_w, const float* __restrict__ out_w,
    const int* __restrict__ counts, const int* __restrict__ tok_list,
    const float* __restrict__ w_list, float* __restrict__ out) {
    int e = blockIdx.x / TILES_PER_E;
    int tile = blockIdx.x % TILES_PER_E;
    int count = counts[e];
    if (tile * TILE_M >= count) return;  // uniform per block

    extern __shared__ char smem[];
    bf16_t* xs = (bf16_t*)smem;                                       // [32][1032]
    bf16_t* hd = (bf16_t*)(smem + TILE_M * XS_STRIDE * 2);            // [32][2056]
    float* pv = (float*)(smem + TILE_M * (XS_STRIDE + HD_STRIDE) * 2);
    int* tk = (int*)(pv + TILE_M);

    int tid = threadIdx.x;
    if (tid < TILE_M) {
        int slot = tile * TILE_M + tid;
        int tok = 0;
        float w = 0.f;
        if (slot < count) {
            tok = tok_list[e * TT + slot];
            w = w_list[e * TT + slot];
        }
        tk[tid] = tok;
        pv[tid] = w;
    }
    __syncthreads();

    // Gather 32 token rows of x into LDS as bf16 (coalesced float4 loads).
    for (int q = tid; q < TILE_M * (HH / 4); q += 256) {
        int row = q >> 8;   // HH/4 == 256 float4 per row
        int c4 = q & 255;
        float4 v = *(const float4*)(x + (size_t)tk[row] * HH + c4 * 4);
        bf16_t* d = xs + row * XS_STRIDE + c4 * 4;
        d[0] = (bf16_t)v.x; d[1] = (bf16_t)v.y;
        d[2] = (bf16_t)v.z; d[3] = (bf16_t)v.w;
    }
    __syncthreads();

    int lane = tid & 31;
    int wave = tid >> 5;
    int half = (lane < 16) ? 0 : 8;  // A-frag K-chunk offset AND D-frag row offset
    int nsub = lane & 15;
    int mrow = lane & 15;

    const float* gw = gate_w + (size_t)e * HH * II;
    const float* uw = up_w + (size_t)e * HH * II;
    const float* ow = out_w + (size_t)e * II * HH;

    // ---- Phase 1: gate/up projections + SwiGLU -> hidden LDS (bf16) ----
    // Wave owns 256 columns of I = 16 WMMA n-tiles.
    for (int nt = 0; nt < 16; ++nt) {
        int nb = wave * 256 + nt * 16;
        v8f ag0 = {}, ag1 = {}, au0 = {}, au1 = {};
        for (int kb = 0; kb < HH; kb += 32) {
            const float* gp = gw + (size_t)(kb + (lane >> 4) * 8) * II + nb + (lane & 15);
            const float* up = uw + (size_t)(kb + (lane >> 4) * 8) * II + nb + (lane & 15);
            __builtin_prefetch(gp + 32 * II, 0, 0);
            __builtin_prefetch(up + 32 * II, 0, 0);
            v16bf bg = load_b_frag(gp, II);
            v16bf bu = load_b_frag(up, II);
            v16bf a0 = load_a_frag(xs, XS_STRIDE, mrow, kb, half);
            v16bf a1 = load_a_frag(xs, XS_STRIDE, 16 + mrow, kb, half);
            ag0 = wmma_bf16(a0, bg, ag0);
            ag1 = wmma_bf16(a1, bg, ag1);
            au0 = wmma_bf16(a0, bu, au0);
            au1 = wmma_bf16(a1, bu, au1);
        }
#pragma unroll
        for (int r = 0; r < 8; ++r) {
            float h0 = silu(ag0[r]) * au0[r];
            float h1 = silu(ag1[r]) * au1[r];
            hd[(half + r) * HD_STRIDE + nb + nsub] = (bf16_t)h0;
            hd[(16 + half + r) * HD_STRIDE + nb + nsub] = (bf16_t)h1;
        }
    }
    __syncthreads();

    // ---- Phase 2: down projection, scale by routing weight, atomic combine ----
    // Wave owns 128 columns of H = 8 WMMA n-tiles.
    for (int nt = 0; nt < 8; ++nt) {
        int nb = wave * 128 + nt * 16;
        v8f ac0 = {}, ac1 = {};
        for (int kb = 0; kb < II; kb += 32) {
            const float* op = ow + (size_t)(kb + (lane >> 4) * 8) * HH + nb + (lane & 15);
            __builtin_prefetch(op + 32 * HH, 0, 0);
            v16bf bo = load_b_frag(op, HH);
            v16bf a0 = load_a_frag(hd, HD_STRIDE, mrow, kb, half);
            v16bf a1 = load_a_frag(hd, HD_STRIDE, 16 + mrow, kb, half);
            ac0 = wmma_bf16(a0, bo, ac0);
            ac1 = wmma_bf16(a1, bo, ac1);
        }
#pragma unroll
        for (int r = 0; r < 8; ++r) {
            int row0 = half + r;
            int row1 = 16 + half + r;
            float v0 = ac0[r] * pv[row0];
            float v1 = ac1[r] * pv[row1];
            atomicAdd(out + (size_t)tk[row0] * HH + nb + nsub, v0);
            atomicAdd(out + (size_t)tk[row1] * HH + nb + nsub, v1);
        }
    }
}

extern "C" void kernel_launch(void* const* d_in, const int* in_sizes, int n_in,
                              void* d_out, int out_size, void* d_ws, size_t ws_size,
                              hipStream_t stream) {
    (void)in_sizes; (void)n_in; (void)out_size; (void)ws_size;
    const float* x = (const float*)d_in[0];        // [B,S,H]
    const float* router_w = (const float*)d_in[1]; // [H,E]
    const float* gate_w = (const float*)d_in[2];   // [E,H,I]
    const float* up_w = (const float*)d_in[3];     // [E,H,I]
    const float* out_w = (const float*)d_in[4];    // [E,I,H]

    float* out = (float*)d_out;             // [T,H]
    float* logits = out + (size_t)TT * HH;  // [T,E]

    // Workspace layout
    int* counts = (int*)d_ws;                                 // [E]
    int* tok_list = (int*)((char*)d_ws + 64);                 // [E][T]
    float* w_list = (float*)((char*)d_ws + 64 + EE * TT * 4); // [E][T]

    // 1) zero output + counters (T*H/4 float4 elements)
    zero_kernel<<<(TT * HH / 4) / 256, 256, 0, stream>>>(out, counts);

    // 2) router: one wave per token
    router_kernel<<<TT / 8, 256, 0, stream>>>(x, router_w, logits, counts,
                                              tok_list, w_list);

    // 3) fused expert FFN over routed tiles
    size_t smem = (size_t)TILE_M * (XS_STRIDE + HD_STRIDE) * 2 + TILE_M * 8;
    moe_expert_kernel<<<EE * TILES_PER_E, 256, smem, stream>>>(
        x, gate_w, up_w, out_w, counts, tok_list, w_list, out);
}
